// DepthMultiPathExecutor_31413390803235
// MI455X (gfx1250) — matrix operand order repaired and checked by reading, not compiled, not run on hardware
//
#include <hip/hip_runtime.h>
#include <hip/hip_bf16.h>
#include <math.h>

#define TD      768
#define HID     128
#define FFN_DIM 512
#define TM      128
#define SFP     132   // f32 LDS row stride (padded, rows 16B-aligned, banks skewed by 4)
#define SHP     136   // f16 LDS row stride (padded, rows 16B-aligned)

typedef __attribute__((ext_vector_type(16))) _Float16 v16h;
typedef __attribute__((ext_vector_type(8)))  _Float16 v8h;
typedef __attribute__((ext_vector_type(4)))  _Float16 v4h;
typedef __attribute__((ext_vector_type(8)))  float    v8f;
typedef __attribute__((ext_vector_type(4)))  float    v4f;

static __device__ __forceinline__ v8f vzero8f() {
  v8f v;
  #pragma unroll
  for (int i = 0; i < 8; ++i) v[i] = 0.0f;
  return v;
}

static __device__ __forceinline__ float gelu_exact(float v) {
  return 0.5f * v * (1.0f + erff(v * 0.70710678118654752440f));
}

// A fragment: 16x32 f16 tile for this wave's 16-row strip.
// ISA layout: lanes 0-15 (row m): K 0-7 in V0-3, K 16-23 in V4-7;
//             lanes 16-31 (row m): K 8-15 in V0-3, K 24-31 in V4-7.
static __device__ __forceinline__ v16h load_a_frag(const _Float16* base, int strip,
                                                   int lane, int kbase) {
  const int m = lane & 15, hi = lane >> 4;
  const _Float16* row = base + (strip + m) * SHP + kbase + hi * 8;
  v8h lo = *(const v8h*)(row);
  v8h h8 = *(const v8h*)(row + 16);
  v16h a;
  #pragma unroll
  for (int i = 0; i < 8; ++i) { a[i] = lo[i]; a[8 + i] = h8[i]; }
  return a;
}

// B fragment: 32x16 f16 (KxN). Weights staged transposed in LDS as wb[n][k],
// so lane n reads K like A: lanes 0-15 K 0-7,16-23; lanes 16-31 K 8-15,24-31.
static __device__ __forceinline__ v16h load_b_frag(const _Float16* wb, int nbase,
                                                   int lane, int kbase) {
  const int n = nbase + (lane & 15), hi = lane >> 4;
  const _Float16* row = wb + n * SHP + kbase + hi * 8;
  v8h x0 = *(const v8h*)(row);
  v8h x1 = *(const v8h*)(row + 16);
  v16h b;
  #pragma unroll
  for (int i = 0; i < 8; ++i) { b[i] = x0[i]; b[8 + i] = x1[i]; }
  return b;
}

#define WMMA_F16(a, b, c) \
  __builtin_amdgcn_wmma_f32_16x16x32_f16(false, (a), false, (b), (short)0, (c), false, false)

// C += A(128x128) * B(128x128), per wave one 16-row strip x 8 N-tiles.
// Batched issue: 4 B fragments loaded ahead, then each WMMA alternates with the
// load of the fragment 4 slots later -> distinct SSA values, loads issued well
// before their consuming WMMA so partial s_wait_dscnt waits are possible.
static __device__ __forceinline__ void gemm128(const _Float16* aBase, const _Float16* wb,
                                               int strip, int lane, v8f c[8]) {
  #pragma unroll
  for (int ks = 0; ks < 4; ++ks) {
    const int kb = ks * 32;
    v16h a  = load_a_frag(aBase, strip, lane, kb);
    v16h b0 = load_b_frag(wb, 0 * 16, lane, kb);
    v16h b1 = load_b_frag(wb, 1 * 16, lane, kb);
    v16h b2 = load_b_frag(wb, 2 * 16, lane, kb);
    v16h b3 = load_b_frag(wb, 3 * 16, lane, kb);
    c[0] = WMMA_F16(a, b0, c[0]);
    v16h b4 = load_b_frag(wb, 4 * 16, lane, kb);
    c[1] = WMMA_F16(a, b1, c[1]);
    v16h b5 = load_b_frag(wb, 5 * 16, lane, kb);
    c[2] = WMMA_F16(a, b2, c[2]);
    v16h b6 = load_b_frag(wb, 6 * 16, lane, kb);
    c[3] = WMMA_F16(a, b3, c[3]);
    v16h b7 = load_b_frag(wb, 7 * 16, lane, kb);
    c[4] = WMMA_F16(a, b4, c[4]);
    c[5] = WMMA_F16(a, b5, c[5]);
    c[6] = WMMA_F16(a, b6, c[6]);
    c[7] = WMMA_F16(a, b7, c[7]);
  }
}

// Warm L0/L2 for an upcoming 128x128 f32 weight chunk (512 cachelines, 2/thread).
static __device__ __forceinline__ void prefetch_w(const float* __restrict__ W, int ldw,
                                                  int row0, int col0, int tid) {
  #pragma unroll
  for (int i = 0; i < 2; ++i) {
    int line = tid + i * 256;              // 0..511
    int r = line >> 2, o = (line & 3) * 32;
    __builtin_prefetch(&W[(size_t)(row0 + r) * ldw + col0 + o], 0, 1);
  }
}

// Stage a 128x128 weight chunk transposed: wb[n][k] = W[(row0+k)*ldw + col0 + n].
// Each thread: fixed n, 4 contiguous k per iteration -> lane-coalesced global
// b32 loads, vector ds_store_b64 to LDS.
static __device__ __forceinline__ void stage_w_T(const float* __restrict__ W, int ldw,
                                                 int row0, int col0,
                                                 _Float16* wb, int tid) {
  const int n  = tid & 127;
  const int kh = (tid >> 7) * 64;
  const float* src = W + (size_t)row0 * ldw + col0 + n;
  #pragma unroll 4
  for (int j = 0; j < 16; ++j) {
    int k0 = kh + j * 4;
    v4h hv;
    #pragma unroll
    for (int i = 0; i < 4; ++i)
      hv[i] = (_Float16)src[(size_t)(k0 + i) * ldw];
    *(v4h*)(&wb[n * SHP + k0]) = hv;
  }
}

// One residual FFN block core: c3 = gelu(LN(inb) @ w1 + b1) @ w2 (b2 added by caller)
static __device__ __forceinline__ void ffn_core(const float* inb,
    const float* __restrict__ g, const float* __restrict__ bet,
    const float* __restrict__ w1, const float* __restrict__ b1,
    const float* __restrict__ w2,
    _Float16* aL, _Float16* a2, _Float16* wbuf,
    int tid, int lane, int strip, v8f c3[8]) {
  if (tid < TM) {                       // LayerNorm, one token row per thread
    const float* src = inb + tid * SFP;
    float s = 0.f, s2 = 0.f;
    #pragma unroll
    for (int k = 0; k < HID; k += 4) {
      v4f v = *(const v4f*)(src + k);
      #pragma unroll
      for (int i = 0; i < 4; ++i) { s += v[i]; s2 += v[i] * v[i]; }
    }
    float mu  = s * (1.0f / HID);
    float var = s2 * (1.0f / HID) - mu * mu;
    float rs  = rsqrtf(var + 1e-5f);
    _Float16* dst = aL + tid * SHP;
    #pragma unroll
    for (int k = 0; k < HID; k += 4) {
      v4f v = *(const v4f*)(src + k);
      v4h hv;
      #pragma unroll
      for (int i = 0; i < 4; ++i)
        hv[i] = (_Float16)((v[i] - mu) * rs * g[k + i] + bet[k + i]);
      *(v4h*)(dst + k) = hv;
    }
  }
  __syncthreads();
  #pragma unroll
  for (int nt = 0; nt < 8; ++nt) c3[nt] = vzero8f();
  const int n16 = lane & 15, hi = lane >> 4;
  for (int f = 0; f < 4; ++f) {         // 512-wide hidden in 4 chunks of 128
    stage_w_T(w1, FFN_DIM, 0, f * 128, wbuf, tid);
    __syncthreads();
    prefetch_w(w2, HID, f * 128, 0, tid);       // warm w2 chunk during GEMM
    v8f c2[8];
    #pragma unroll
    for (int nt = 0; nt < 8; ++nt) c2[nt] = vzero8f();
    gemm128(aL, wbuf, strip, lane, c2);
    #pragma unroll
    for (int nt = 0; nt < 8; ++nt) {    // bias + exact-erf GELU -> f16 stage
      int n = nt * 16 + n16;
      float bias = b1[f * 128 + n];
      #pragma unroll
      for (int r = 0; r < 8; ++r) {
        int row = strip + hi * 8 + r;
        a2[row * SHP + n] = (_Float16)gelu_exact(c2[nt][r] + bias);
      }
    }
    __syncthreads();
    stage_w_T(w2, HID, f * 128, 0, wbuf, tid);
    __syncthreads();
    if (f < 3)
      prefetch_w(w1, FFN_DIM, 0, (f + 1) * 128, tid); // warm next w1 chunk
    gemm128(a2, wbuf, strip, lane, c3); // accumulate h@w2 across chunks
    __syncthreads();
  }
}

__global__ __launch_bounds__(256, 1)
void dmpe_fused(const float* __restrict__ toks, const float* __restrict__ rp,
                const float* __restrict__ w_in, const float* __restrict__ b_in,
                const float* __restrict__ sh_g, const float* __restrict__ sh_b,
                const float* __restrict__ sh_w1, const float* __restrict__ sh_b1,
                const float* __restrict__ sh_w2, const float* __restrict__ sh_b2,
                const float* __restrict__ dp_g, const float* __restrict__ dp_b,
                const float* __restrict__ dp_w1, const float* __restrict__ dp_b1,
                const float* __restrict__ dp_w2, const float* __restrict__ dp_b2,
                const float* __restrict__ w_out, const float* __restrict__ b_out,
                float* __restrict__ out, float* __restrict__ stats, float inv_ntok) {
  extern __shared__ char smem[];
  float*    xbuf = (float*)smem;                    // skip activation x
  float*    hbuf = xbuf + TM * SFP;                 // deep-path running h
  float*    fbuf = hbuf + TM * SFP;                 // fused accumulator
  _Float16* aL   = (_Float16*)(fbuf + TM * SFP);    // A-operand staging
  _Float16* a2   = aL + TM * SHP;                   // gelu-chunk staging
  _Float16* wbuf = a2 + TM * SHP;                   // transposed weight staging

  const int tid = threadIdx.x;
  const int lane = tid & 31;
  const int strip = (tid >> 5) * 16;
  const int n16 = lane & 15, hi = lane >> 4;
  const int gtok0 = blockIdx.x * TM;

  // ---- Phase A: x = tokens @ w_in + b_in (K=768 in 6 chunks) ----
  v8f cin[8];
  #pragma unroll
  for (int nt = 0; nt < 8; ++nt) cin[nt] = vzero8f();
  for (int kc = 0; kc < 6; ++kc) {
    #pragma unroll 4
    for (int i = 0; i < 16; ++i) {      // vectorized: b128 global -> b64 LDS
      int gi = tid + i * 256;
      int row = gi >> 5, col = (gi & 31) * 4;
      v4f v = *(const v4f*)(&toks[(size_t)(gtok0 + row) * TD + kc * 128 + col]);
      v4h hv;
      #pragma unroll
      for (int q = 0; q < 4; ++q) hv[q] = (_Float16)v[q];
      *(v4h*)(&aL[row * SHP + col]) = hv;
    }
    stage_w_T(w_in, HID, kc * 128, 0, wbuf, tid);
    __syncthreads();
    if (kc < 5)
      prefetch_w(w_in, HID, (kc + 1) * 128, 0, tid);
    gemm128(aL, wbuf, strip, lane, cin);
    __syncthreads();
  }
  #pragma unroll
  for (int r = 0; r < 8; ++r) {
    int row = strip + hi * 8 + r;
    #pragma unroll
    for (int nt = 0; nt < 8; ++nt) {
      int n = nt * 16 + n16;
      float v = cin[nt][r] + b_in[n];
      xbuf[row * SFP + n] = v;
      hbuf[row * SFP + n] = v;
    }
  }
  __syncthreads();

  // ---- Phase B: shallow block; fused = (p0+p1)*x + p1*ffn(x) ----
  v8f c3[8];
  ffn_core(xbuf, sh_g, sh_b, sh_w1, sh_b1, sh_w2, aL, a2, wbuf, tid, lane, strip, c3);
  #pragma unroll
  for (int r = 0; r < 8; ++r) {
    int row = strip + hi * 8 + r;
    float p0 = rp[(size_t)(gtok0 + row) * 3 + 0];
    float p1 = rp[(size_t)(gtok0 + row) * 3 + 1];
    #pragma unroll
    for (int nt = 0; nt < 8; ++nt) {
      int n = nt * 16 + n16;
      float ffnv = c3[nt][r] + sh_b2[n];
      fbuf[row * SFP + n] = (p0 + p1) * xbuf[row * SFP + n] + p1 * ffnv;
    }
  }
  __syncthreads();

  // ---- Phase C: 3 deep blocks; after last, fused += p2 * h ----
  for (int L = 0; L < 3; ++L) {
    ffn_core(hbuf, dp_g + L * HID, dp_b + L * HID,
             dp_w1 + (size_t)L * HID * FFN_DIM, dp_b1 + L * FFN_DIM,
             dp_w2 + (size_t)L * FFN_DIM * HID,
             aL, a2, wbuf, tid, lane, strip, c3);
    const float* b2p = dp_b2 + L * HID;
    #pragma unroll
    for (int r = 0; r < 8; ++r) {
      int row = strip + hi * 8 + r;
      float p2 = (L == 2) ? rp[(size_t)(gtok0 + row) * 3 + 2] : 0.f;
      #pragma unroll
      for (int nt = 0; nt < 8; ++nt) {
        int n = nt * 16 + n16;
        float nh = hbuf[row * SFP + n] + c3[nt][r] + b2p[n];
        hbuf[row * SFP + n] = nh;
        if (L == 2) fbuf[row * SFP + n] += p2 * nh;
      }
    }
    __syncthreads();
  }

  // ---- Phase D: out = fused @ w_out + b_out (N=768 in 6 chunks) ----
  #pragma unroll 4
  for (int i = 0; i < 16; ++i) {        // vectorized f32->f16 restage of fused
    int gi = tid + i * 256;
    int row = gi >> 5, col = (gi & 31) * 4;
    v4f v = *(const v4f*)(&fbuf[row * SFP + col]);
    v4h hv;
    #pragma unroll
    for (int q = 0; q < 4; ++q) hv[q] = (_Float16)v[q];
    *(v4h*)(&aL[row * SHP + col]) = hv;
  }
  __syncthreads();
  for (int nc = 0; nc < 6; ++nc) {
    stage_w_T(w_out, TD, 0, nc * 128, wbuf, tid);
    __syncthreads();
    if (nc < 5)
      prefetch_w(w_out, TD, 0, (nc + 1) * 128, tid);
    v8f co[8];
    #pragma unroll
    for (int nt = 0; nt < 8; ++nt) co[nt] = vzero8f();
    gemm128(aL, wbuf, strip, lane, co);
    #pragma unroll
    for (int r = 0; r < 8; ++r) {
      int row = strip + hi * 8 + r;
      float* orow = out + (size_t)(gtok0 + row) * TD + nc * 128;
      #pragma unroll
      for (int nt = 0; nt < 8; ++nt) {
        int n = nt * 16 + n16;
        orow[n] = co[nt][r] + b_out[nc * 128 + n];
      }
    }
    __syncthreads();
  }

  // ---- route stats ----
  if (tid < TM) {
    int gt = gtok0 + tid;
    #pragma unroll
    for (int j = 0; j < 3; ++j)
      atomicAdd(stats + j, rp[(size_t)gt * 3 + j] * inv_ntok);
  }
}

extern "C" void kernel_launch(void* const* d_in, const int* in_sizes, int n_in,
                              void* d_out, int out_size, void* d_ws, size_t ws_size,
                              hipStream_t stream) {
  (void)n_in; (void)out_size; (void)d_ws; (void)ws_size;
  const float* toks  = (const float*)d_in[0];
  const float* rp    = (const float*)d_in[1];
  const float* w_in  = (const float*)d_in[2];
  const float* b_in  = (const float*)d_in[3];
  const float* sh_g  = (const float*)d_in[4];
  const float* sh_b  = (const float*)d_in[5];
  const float* sh_w1 = (const float*)d_in[6];
  const float* sh_b1 = (const float*)d_in[7];
  const float* sh_w2 = (const float*)d_in[8];
  const float* sh_b2 = (const float*)d_in[9];
  const float* dp_g  = (const float*)d_in[10];
  const float* dp_b  = (const float*)d_in[11];
  const float* dp_w1 = (const float*)d_in[12];
  const float* dp_b1 = (const float*)d_in[13];
  const float* dp_w2 = (const float*)d_in[14];
  const float* dp_b2 = (const float*)d_in[15];
  const float* w_out = (const float*)d_in[16];
  const float* b_out = (const float*)d_in[17];

  const int ntok = in_sizes[0] / TD;           // 65536
  float* out   = (float*)d_out;
  float* stats = out + (size_t)ntok * TD;      // 3 trailing route-stat slots

  hipMemsetAsync(stats, 0, 3 * sizeof(float), stream);  // capture-safe

  const size_t shmem = (size_t)(3 * TM * SFP) * sizeof(float)
                     + (size_t)(3 * TM * SHP) * sizeof(_Float16);  // 300 KB
  hipFuncSetAttribute((const void*)dmpe_fused,
                      hipFuncAttributeMaxDynamicSharedMemorySize, (int)shmem);

  dim3 grid(ntok / TM), block(256);
  dmpe_fused<<<grid, block, shmem, stream>>>(
      toks, rp, w_in, b_in, sh_g, sh_b, sh_w1, sh_b1, sh_w2, sh_b2,
      dp_g, dp_b, dp_w1, dp_b1, dp_w2, dp_b2, w_out, b_out,
      out, stats, 1.0f / (float)ntok);
}
